// CustomDetector_53515292508715
// MI455X (gfx1250) — compile-verified, matrix operand order checked by reading, not run on hardware
//
#include <hip/hip_runtime.h>
#include <hip/hip_bf16.h>
#include <math.h>
#include <stdint.h>

typedef __attribute__((ext_vector_type(16))) __bf16 v16bf;
typedef __attribute__((ext_vector_type(8)))  float  v8f;

#define TILE_M 128
#define TILE_N 64
#define LDSTR  40                 // bf16 per LDS row (32 used + 8 pad -> 80B, 16B aligned)
#define AELEMS (TILE_M * LDSTR)   // 5120 bf16
#define BELEMS (TILE_N * LDSTR)   // 2560 bf16
#define USE_ASYNC_LDS 1

struct __attribute__((aligned(16))) U16B { unsigned int x[4]; };
union FragU { U16B u[2]; v16bf v; };

struct ConvArgs {
  const __bf16* srcB;   // padded NHWC bf16 (asrc==0)
  const float*  srcF;   // NCHW fp32      (asrc==1)
  const __bf16* wgt;    // [coPad][K] bf16, K = taps*Cin, tap-major
  const float*  bias;   // [coReal]
  __bf16*       dstB;   // padded NHWC bf16 output (epi bit2 clear)
  float*        dstF;   // final fp32 output base (epi bit2 set)
  const __bf16* up;     // upstream padded NHWC bf16 for upsample-add
  int H, W, Cin, K, coReal;
  int asrc, epi, upH, upW, chOff;
  long long pixOff;     // level pixel offset within [5376]
};

// epi bits: 1 = ReLU, 2 = upsample-add, 4 = final fp32 write to d_out

#if USE_ASYNC_LDS
__device__ __forceinline__ void async_ld_b128(const void* g, void* l) {
  unsigned lds = (unsigned)(uintptr_t)l;                 // low 32b of flat shared addr = LDS addr
  unsigned long long ga = (unsigned long long)(uintptr_t)g;
  asm volatile("global_load_async_to_lds_b128 %0, %1, off"
               :: "v"(lds), "v"(ga) : "memory");
}
__device__ __forceinline__ void wait_async() {
  asm volatile("s_wait_asynccnt 0x0" ::: "memory");
}
#endif

__global__ __launch_bounds__(256) void conv_gemm(ConvArgs a) {
  __shared__ __bf16 sA[2 * AELEMS];
  __shared__ __bf16 sB[2 * BELEMS];

  const int tid    = threadIdx.x;
  const int lane   = tid & 31;
  const int waveId = tid >> 5;
  const int wm     = waveId & 3;   // 4 m-waves of 32 rows
  const int wn     = waveId >> 2;  // 2 n-waves of 32 cols
  const int lrow   = lane & 15;
  const int lsel   = lane >> 4;
  const int HW     = a.H * a.W;
  const int mBase  = blockIdx.x * TILE_M;
  const int nBase  = blockIdx.y * TILE_N;
  const int Wp     = a.W + 2;

  // ---- loop-invariant staging geometry ----
  const int rB  = tid >> 2;                 // weight row
  const int ccB = (tid & 3) << 3;
  const __bf16* gB = a.wgt + (long long)(nBase + rB) * a.K + ccB;
  const int sBoff = rB * LDSTR + ccB;

  long long aCen[2]; int sAoff[2];
  const float* gF = nullptr; int sFoff = 0;
  if (a.asrc == 0) {
#pragma unroll
    for (int it = 0; it < 2; ++it) {
      int idx = it * 256 + tid;             // 512 chunks of 16B
      int p = idx >> 2, cc = (idx & 3) << 3;
      int P = mBase + p;
      int b = P / HW; int hw = P - b * HW;
      int h = hw / a.W; int w = hw - h * a.W;
      aCen[it] = ((long long)(b * (a.H + 2) + h + 1) * Wp + (w + 1)) * 256 + cc;
      sAoff[it] = p * LDSTR + cc;
    }
  } else {
    int p  = tid & 127;
    int kh = (tid >> 7) << 4;               // 0 or 16: 16 consecutive k per thread
    int P  = mBase + p;
    int b  = P / HW; int hw = P - b * HW;
    gF = a.srcF + (long long)(b * a.Cin + kh) * HW + hw;
    sFoff = p * LDSTR + kh;
  }

  int fAoff[2], fBoff[2];
#pragma unroll
  for (int f = 0; f < 2; ++f) {
    fAoff[f] = (wm * 32 + f * 16 + lrow) * LDSTR + lsel * 8;
    fBoff[f] = (wn * 32 + f * 16 + lrow) * LDSTR + lsel * 8;
  }

  v8f acc[2][2];
#pragma unroll
  for (int i = 0; i < 2; ++i)
#pragma unroll
    for (int j = 0; j < 2; ++j)
#pragma unroll
      for (int e = 0; e < 8; ++e) acc[i][j][e] = 0.f;

  auto compute = [&](int bsel) {
    const __bf16* cA = sA + bsel * AELEMS;
    const __bf16* cB = sB + bsel * BELEMS;
    FragU fa[2], fb[2];
#pragma unroll
    for (int f = 0; f < 2; ++f) {
      fa[f].u[0] = *(const U16B*)&cA[fAoff[f]];
      fa[f].u[1] = *(const U16B*)&cA[fAoff[f] + 16];
      fb[f].u[0] = *(const U16B*)&cB[fBoff[f]];
      fb[f].u[1] = *(const U16B*)&cB[fBoff[f] + 16];
    }
#pragma unroll
    for (int i = 0; i < 2; ++i)
#pragma unroll
      for (int j = 0; j < 2; ++j)
        acc[i][j] = __builtin_amdgcn_wmma_f32_16x16x32_bf16(
            false, fa[i].v, false, fb[j].v, (short)0, acc[i][j], false, false);
  };

  const int ksteps = a.K >> 5;

#if USE_ASYNC_LDS
  if (a.asrc == 0) {
    // CDNA5 async global->LDS (ASYNCcnt), double-buffered
    auto issueA = [&](int ks, int bsel) {
      const int k0 = ks << 5;
      const int t = k0 >> 8, ci0 = k0 & 255;
      const int dy = t / 3 - 1, dx = t - (t / 3) * 3 - 1;
      const long long doff = (long long)(dy * Wp + dx) * 256 + ci0;
      __bf16* dA = sA + bsel * AELEMS;
      __bf16* dB = sB + bsel * BELEMS;
      async_ld_b128(a.srcB + aCen[0] + doff, &dA[sAoff[0]]);
      async_ld_b128(a.srcB + aCen[1] + doff, &dA[sAoff[1]]);
      async_ld_b128(gB + k0, &dB[sBoff]);
    };
    issueA(0, 0);
    wait_async();
    __syncthreads();
    for (int ks = 0; ks < ksteps; ++ks) {
      const int cur = ks & 1;
      if (ks + 1 < ksteps) issueA(ks + 1, cur ^ 1);
      compute(cur);
      wait_async();
      __syncthreads();
    }
  } else
#endif
  {
    // sync path (register prefetch, double-buffered)
    U16B pa0, pa1, pb; float pf[16];
    auto issueS = [&](int ks) {
      const int k0 = ks << 5;
      pb = *(const U16B*)(gB + k0);
      if (a.asrc == 0) {
        const int t = k0 >> 8, ci0 = k0 & 255;
        const int dy = t / 3 - 1, dx = t - (t / 3) * 3 - 1;
        const long long doff = (long long)(dy * Wp + dx) * 256 + ci0;
        pa0 = *(const U16B*)(a.srcB + aCen[0] + doff);
        pa1 = *(const U16B*)(a.srcB + aCen[1] + doff);
      } else {
        const float* g = gF + (long long)k0 * HW;
#pragma unroll
        for (int j = 0; j < 16; ++j) pf[j] = g[(long long)j * HW];
      }
    };
    auto storeS = [&](int bsel) {
      __bf16* dA = sA + bsel * AELEMS;
      __bf16* dB = sB + bsel * BELEMS;
      *(U16B*)&dB[sBoff] = pb;
      if (a.asrc == 0) {
        *(U16B*)&dA[sAoff[0]] = pa0;
        *(U16B*)&dA[sAoff[1]] = pa1;
      } else {
#pragma unroll
        for (int j = 0; j < 16; j += 2) {
          union { __bf16 h[2]; unsigned int u; } pk;
          pk.h[0] = (__bf16)pf[j];
          pk.h[1] = (__bf16)pf[j + 1];
          *(unsigned int*)&dA[sFoff + j] = pk.u;
        }
      }
    };
    issueS(0);
    storeS(0);
    __syncthreads();
    for (int ks = 0; ks < ksteps; ++ks) {
      const int cur = ks & 1;
      if (ks + 1 < ksteps) issueS(ks + 1);
      compute(cur);
      if (ks + 1 < ksteps) storeS(cur ^ 1);
      __syncthreads();
    }
  }

  // Epilogue: C/D layout -> VGPR e holds (M = e + lsel*8, N = lrow)
#pragma unroll
  for (int i = 0; i < 2; ++i) {
#pragma unroll
    for (int j = 0; j < 2; ++j) {
#pragma unroll
      for (int e = 0; e < 8; ++e) {
        int gm = mBase + wm * 32 + i * 16 + e + lsel * 8;
        int co = nBase + wn * 32 + j * 16 + lrow;
        float val = acc[i][j][e];
        if (co < a.coReal) val += a.bias[co];
        int b = gm / HW; int hw = gm - b * HW;
        int h = hw / a.W; int w = hw - h * a.W;
        if (a.epi & 2) {
          val += (float)a.up[(((long long)b * (a.upH + 2) + ((h >> 1) + 1)) * (a.upW + 2) +
                              ((w >> 1) + 1)) * 256 + co];
        }
        if (a.epi & 1) val = fmaxf(val, 0.f);
        if (a.epi & 4) {
          if (co < a.coReal)
            a.dstF[((long long)b * 5376 + a.pixOff + hw) * 252 + a.chOff + co] = val;
        } else {
          a.dstB[(((long long)b * (a.H + 2) + (h + 1)) * Wp + (w + 1)) * 256 + co] = (__bf16)val;
        }
      }
    }
  }
}

// Pack fp32 (O,I,kh,kw) weights into bf16 [coPad][t*I + ci], t = ky*3+kx (or t=0 for 1x1)
__global__ void pack_weights(const float* __restrict__ w, __bf16* __restrict__ dst,
                             int O, int I, int taps, int coPad) {
  long long idx   = (long long)blockIdx.x * blockDim.x + threadIdx.x;
  long long total = (long long)coPad * taps * I;
  if (idx >= total) return;
  int TI = taps * I;
  int co = (int)(idx / TI);
  int rem = (int)(idx - (long long)co * TI);
  int t  = rem / I;
  int ci = rem - t * I;
  float v = 0.f;
  if (co < O) v = w[((long long)co * I + ci) * taps + t];
  dst[(long long)co * TI + t * I + ci] = (__bf16)v;
}

__global__ void zero_u16(U16B* p, long long n16) {
  long long i      = (long long)blockIdx.x * blockDim.x + threadIdx.x;
  long long stride = (long long)gridDim.x * blockDim.x;
  U16B z; z.x[0] = 0u; z.x[1] = 0u; z.x[2] = 0u; z.x[3] = 0u;
  for (; i < n16; i += stride) p[i] = z;
}

__global__ void anchors_kernel(float* __restrict__ out) {
  int a = blockIdx.x * blockDim.x + threadIdx.x;
  if (a >= 16128) return;
  int off, Wl; float bs, stride;
  if (a < 12288)      { off = 0;     Wl = 64; bs = 32.f;  stride = 8.f;  }
  else if (a < 15360) { off = 12288; Wl = 32; bs = 64.f;  stride = 16.f; }
  else                { off = 15360; Wl = 16; bs = 128.f; stride = 32.f; }
  int i  = a - off;
  int px = i / 3;
  int ar = i - px * 3;
  float arv = (ar == 0) ? 0.5f : ((ar == 1) ? 1.0f : 2.0f);
  float sa  = sqrtf(arv);
  float hx  = bs * sa * 0.5f;
  float hy  = bs / sa * 0.5f;
  float x = (float)(px % Wl) * stride;
  float y = (float)(px / Wl) * stride;
  out[a * 4 + 0] = x - hx;
  out[a * 4 + 1] = y - hy;
  out[a * 4 + 2] = x + hx;
  out[a * 4 + 3] = y + hy;
}

static void launch_conv(hipStream_t s, const ConvArgs& a, int coPad) {
  int Mtiles = (8 * a.H * a.W) / TILE_M;   // H*W in {4096,1024,256}: divisible
  dim3 grid(Mtiles, coPad / TILE_N);
  conv_gemm<<<grid, dim3(256), 0, s>>>(a);
}

extern "C" void kernel_launch(void* const* d_in, const int* in_sizes, int n_in,
                              void* d_out, int out_size, void* d_ws, size_t ws_size,
                              hipStream_t stream) {
  (void)in_sizes; (void)n_in; (void)out_size; (void)ws_size;
  const float* feat3 = (const float*)d_in[0];
  const float* feat4 = (const float*)d_in[1];
  const float* feat5 = (const float*)d_in[2];
  const float* lw0 = (const float*)d_in[3];  const float* lb0 = (const float*)d_in[4];
  const float* lw1 = (const float*)d_in[5];  const float* lb1 = (const float*)d_in[6];
  const float* lw2 = (const float*)d_in[7];  const float* lb2 = (const float*)d_in[8];
  const float* fw0 = (const float*)d_in[9];  const float* fb0 = (const float*)d_in[10];
  const float* fw1 = (const float*)d_in[11]; const float* fb1 = (const float*)d_in[12];
  const float* fw2 = (const float*)d_in[13]; const float* fb2 = (const float*)d_in[14];
  const float* cw0 = (const float*)d_in[15]; const float* cb0 = (const float*)d_in[16];
  const float* cw1 = (const float*)d_in[17]; const float* cb1 = (const float*)d_in[18];
  const float* cw2 = (const float*)d_in[19]; const float* cb2 = (const float*)d_in[20];
  const float* rw0 = (const float*)d_in[21]; const float* rb0 = (const float*)d_in[22];
  const float* rw1 = (const float*)d_in[23]; const float* rb1 = (const float*)d_in[24];
  const float* rw2 = (const float*)d_in[25]; const float* rb2 = (const float*)d_in[26];

  char*  ws  = (char*)d_ws;
  size_t off = 0;
  auto alloc = [&](size_t bytes) -> char* {
    char* p = ws + off;
    off = (off + bytes + 255) & ~(size_t)255;
    return p;
  };

  // ---- packed weights (bf16) ----
  __bf16* wL0 = (__bf16*)alloc((size_t)256 * 512  * 2);
  __bf16* wL1 = (__bf16*)alloc((size_t)256 * 1024 * 2);
  __bf16* wL2 = (__bf16*)alloc((size_t)256 * 2048 * 2);
  __bf16* wF[3];
  for (int l = 0; l < 3; ++l) wF[l] = (__bf16*)alloc((size_t)256 * 2304 * 2);
  __bf16* wC0 = (__bf16*)alloc((size_t)256 * 2304 * 2);
  __bf16* wC1 = (__bf16*)alloc((size_t)256 * 2304 * 2);
  __bf16* wC2 = (__bf16*)alloc((size_t)256 * 2304 * 2);   // 240 real, padded to 256
  __bf16* wR0 = (__bf16*)alloc((size_t)256 * 2304 * 2);
  __bf16* wR1 = (__bf16*)alloc((size_t)256 * 2304 * 2);
  __bf16* wR2 = (__bf16*)alloc((size_t)64  * 2304 * 2);   // 12 real, padded to 64

  // ---- padded NHWC bf16 buffers (zero halo) ----
  const int Hs[3] = {64, 32, 16};
  __bf16 *L[3], *P[3], *Ta[3], *Tb[3];
  char* padStart = ws + off;
  for (int l = 0; l < 3; ++l) {
    size_t bytes = (size_t)8 * (Hs[l] + 2) * (Hs[l] + 2) * 256 * 2;
    L[l]  = (__bf16*)alloc(bytes);
    P[l]  = (__bf16*)alloc(bytes);
    Ta[l] = (__bf16*)alloc(bytes);
    Tb[l] = (__bf16*)alloc(bytes);
  }
  size_t padBytes = (size_t)((ws + off) - padStart);

  // zero halos (zero whole padded region; interiors get overwritten)
  {
    long long n16 = (long long)(padBytes / 16);
    int blocks = (int)((n16 + 255) / 256);
    if (blocks > 4096) blocks = 4096;
    zero_u16<<<blocks, 256, 0, stream>>>((U16B*)padStart, n16);
  }

  auto pack = [&](const float* w, __bf16* dst, int O, int I, int taps, int coPad) {
    long long total = (long long)coPad * taps * I;
    int blocks = (int)((total + 255) / 256);
    pack_weights<<<blocks, 256, 0, stream>>>(w, dst, O, I, taps, coPad);
  };
  pack(lw0, wL0, 256, 512,  1, 256);
  pack(lw1, wL1, 256, 1024, 1, 256);
  pack(lw2, wL2, 256, 2048, 1, 256);
  pack(fw0, wF[0], 256, 256, 9, 256);
  pack(fw1, wF[1], 256, 256, 9, 256);
  pack(fw2, wF[2], 256, 256, 9, 256);
  pack(cw0, wC0, 256, 256, 9, 256);
  pack(cw1, wC1, 256, 256, 9, 256);
  pack(cw2, wC2, 240, 256, 9, 256);
  pack(rw0, wR0, 256, 256, 9, 256);
  pack(rw1, wR1, 256, 256, 9, 256);
  pack(rw2, wR2, 12,  256, 9, 64);

  // ---- laterals (1x1, fp32 NCHW source) top-down with fused upsample-add ----
  ConvArgs a = {};
  a.srcF = feat5; a.wgt = wL2; a.bias = lb2; a.dstB = L[2];
  a.H = 16; a.W = 16; a.Cin = 2048; a.K = 2048; a.coReal = 256;
  a.asrc = 1; a.epi = 0;
  launch_conv(stream, a, 256);
  a.srcF = feat4; a.wgt = wL1; a.bias = lb1; a.dstB = L[1];
  a.H = 32; a.W = 32; a.Cin = 1024; a.K = 1024;
  a.epi = 2; a.up = L[2]; a.upH = 16; a.upW = 16;
  launch_conv(stream, a, 256);
  a.srcF = feat3; a.wgt = wL0; a.bias = lb0; a.dstB = L[0];
  a.H = 64; a.W = 64; a.Cin = 512; a.K = 512;
  a.epi = 2; a.up = L[1]; a.upH = 32; a.upW = 32;
  launch_conv(stream, a, 256);

  // ---- FPN 3x3 output convs ----
  const float* fb[3] = {fb0, fb1, fb2};
  for (int l = 0; l < 3; ++l) {
    ConvArgs c = {};
    c.srcB = L[l]; c.wgt = wF[l]; c.bias = fb[l]; c.dstB = P[l];
    c.H = Hs[l]; c.W = Hs[l]; c.Cin = 256; c.K = 2304; c.coReal = 256;
    c.asrc = 0; c.epi = 0;
    launch_conv(stream, c, 256);
  }

  // ---- towers (shared weights), final convs write straight into d_out NHWC ----
  const long long pixOff[3] = {0, 4096, 5120};
  float* outF = (float*)d_out;
  for (int l = 0; l < 3; ++l) {
    ConvArgs c = {};
    c.H = Hs[l]; c.W = Hs[l]; c.Cin = 256; c.K = 2304; c.asrc = 0;
    // cls tower
    c.srcB = P[l];  c.wgt = wC0; c.bias = cb0; c.dstB = Ta[l]; c.coReal = 256; c.epi = 1;
    launch_conv(stream, c, 256);
    c.srcB = Ta[l]; c.wgt = wC1; c.bias = cb1; c.dstB = Tb[l]; c.coReal = 256; c.epi = 1;
    launch_conv(stream, c, 256);
    c.srcB = Tb[l]; c.wgt = wC2; c.bias = cb2; c.dstF = outF;  c.coReal = 240;
    c.epi = 4; c.chOff = 0; c.pixOff = pixOff[l];
    launch_conv(stream, c, 256);
    // reg tower (reuse Ta/Tb; same-size interiors keep halos zero)
    c.epi = 1; c.chOff = 0; c.pixOff = 0;
    c.srcB = P[l];  c.wgt = wR0; c.bias = rb0; c.dstB = Ta[l]; c.coReal = 256;
    launch_conv(stream, c, 256);
    c.srcB = Ta[l]; c.wgt = wR1; c.bias = rb1; c.dstB = Tb[l]; c.coReal = 256;
    launch_conv(stream, c, 256);
    c.srcB = Tb[l]; c.wgt = wR2; c.bias = rb2; c.dstF = outF;  c.coReal = 12;
    c.epi = 4; c.chOff = 240; c.pixOff = pixOff[l];
    launch_conv(stream, c, 64);
  }

  // ---- anchors appended after [8,5376,252] ----
  float* anch = outF + (long long)8 * 5376 * 252;
  anchors_kernel<<<(16128 + 255) / 256, 256, 0, stream>>>(anch);
}
